// MultiQueryAttention_59768764891206
// MI455X (gfx1250) — matrix operand order repaired and checked by reading, not compiled, not run on hardware
//
#include <hip/hip_runtime.h>
#include <hip/hip_bf16.h>
#include <stdint.h>

// ---------- types ----------
typedef __attribute__((ext_vector_type(16))) __bf16 v16bf;
typedef __attribute__((ext_vector_type(8)))  __bf16 v8bf;
typedef __attribute__((ext_vector_type(8)))  float  v8f;
typedef __attribute__((ext_vector_type(8)))  float  f8;
typedef __attribute__((ext_vector_type(4)))  unsigned int u32x4;
typedef __attribute__((ext_vector_type(8)))  int i32x8;
typedef __attribute__((ext_vector_type(4)))  int i32x4;

#define Bsz 2
#define Ssz 1024
#define Esz 1024
#define Hsz 8
#define HD  128
#define SH  (Ssz * Hsz)      // 8192
#define RTOT (Bsz * Ssz)     // 2048

// LDS A-tile: 64 rows x 128 K bf16, TDM-padded +16B per 256B row -> 136 elems/row
#define PV_KCH   128
#define PV_ROWS  64
#define PV_LROW  136

// ---------- helpers ----------
__device__ __forceinline__ __bf16 f2bf(float f) { return (__bf16)f; }

// A-fragment (16x32 bf16, MxK): lane m = l&15; elements 0..15 -> K = 16*(l>>4)+0..15
// (contiguous order, matching the B fragments)
__device__ __forceinline__ v16bf load_a_frag(const __bf16* row_ptr, int lane) {
  const int base0 = (lane >> 4) * 16;
  v8bf lo = *(const v8bf*)(row_ptr + base0);
  v8bf hi = *(const v8bf*)(row_ptr + 8 + base0);
  return __builtin_shufflevector(lo, hi, 0,1,2,3,4,5,6,7,8,9,10,11,12,13,14,15);
}

// Same A-fragment but source is fp32 (converted in-register)
__device__ __forceinline__ v16bf load_a_frag_f32(const float* row_ptr, int lane) {
  const int base0 = (lane >> 4) * 16;
  f8 lo = *(const f8*)(row_ptr + base0);
  f8 hi = *(const f8*)(row_ptr + 8 + base0);
  v16bf a;
#pragma unroll
  for (int i = 0; i < 8; ++i) { a[i] = f2bf(lo[i]); a[8 + i] = f2bf(hi[i]); }
  return a;
}

__device__ __forceinline__ v8f wmma_bf16(v16bf a, v16bf b, v8f c) {
  return __builtin_amdgcn_wmma_f32_16x16x32_bf16(false, a, false, b, (short)0, c,
                                                 false, false);
}

// Issue one TDM load: 2D bf16 tile [PV_KCH x PV_ROWS] from a [1024 x 8192] matrix
// (row stride 8192 elems) into LDS at lds_off, with per-row padding of 4 DWORDs
// every 64 DWORDs (so LDS row stride = 272B, bank-conflict-free fragment reads).
__device__ __forceinline__ void tdm_load_a(unsigned int lds_off,
                                           const __bf16* gaddr) {
  const unsigned long long ga = (unsigned long long)(uintptr_t)gaddr;
  u32x4 g0;
  g0[0] = 1u;                                    // count=1, user mode
  g0[1] = lds_off;                               // lds_addr
  g0[2] = (unsigned int)ga;                      // global_addr[31:0]
  g0[3] = (unsigned int)((ga >> 32) & 0x01FFFFFFu) | 0x80000000u; // [56:32]|type=2
  i32x8 g1;
  g1[0] = (int)((1u << 16) |                     // data_size = 2 bytes
                (1u << 20) |                     // pad_enable
                (5u << 22) |                     // pad_interval: every 64 DWORDs
                (3u << 25));                     // pad_amount: 4 DWORDs
  g1[1] = (int)(8192u << 16);                    // tensor_dim0 = 8192 (lo16 in hi half)
  g1[2] = (int)(1024u << 16);                    // tensor_dim1 = 1024
  g1[3] = (int)((unsigned)PV_KCH << 16);         // tile_dim0 = 128
  g1[4] = PV_ROWS;                               // tile_dim1 = 64, tile_dim2 = 0
  g1[5] = 8192;                                  // tensor_dim0_stride (lo32)
  g1[6] = 0;
  g1[7] = 0;
  i32x4 z4 = (i32x4)0;
#if __has_include(<hip/amd_detail/amd_gfx1250_TDM.h>)
  i32x8 z8 = (i32x8)0;
  __builtin_amdgcn_tensor_load_to_lds(g0, g1, z4, z4, z8, 0);
#else
  __builtin_amdgcn_tensor_load_to_lds(g0, g1, z4, z4, 0);
#endif
}

// ---------- kernels ----------

// fp32 -> bf16, 8 elements per thread (b128 in / b128 out per lane pair)
__global__ void cvt_f32_bf16(const float* __restrict__ in, __bf16* __restrict__ out,
                             size_t n8) {
  size_t i = (size_t)blockIdx.x * blockDim.x + threadIdx.x;
  if (i >= n8) return;
  f8 x = *(const f8*)(in + i * 8);
  v8bf y;
#pragma unroll
  for (int j = 0; j < 8; ++j) y[j] = f2bf(x[j]);
  *(v8bf*)(out + i * 8) = y;
}

// C = A(bf16,[MxK]) @ W(bf16,[NxK])^T + bias(f32,[N]); one wave -> 16x64 tile.
template <bool OUT_BF16>
__global__ void __launch_bounds__(256)
gemm_xwT(const __bf16* __restrict__ A, const __bf16* __restrict__ W,
         const float* __restrict__ bias, void* __restrict__ Out,
         int M, int N, int K) {
  const int lane = threadIdx.x & 31;
  const int wave = threadIdx.x >> 5;
  const int row0 = blockIdx.y * 128 + wave * 16;
  const int col0 = blockIdx.x * 64;
  const __bf16* aptr = A + (size_t)(row0 + (lane & 15)) * K;
  const __bf16* bptr[4];
#pragma unroll
  for (int t = 0; t < 4; ++t)
    bptr[t] = W + (size_t)(col0 + t * 16 + (lane & 15)) * K + ((lane >> 4) << 4);

  v8f c[4] = {};
  v16bf a_cur = load_a_frag(aptr, lane);
  v16bf b_cur[4];
#pragma unroll
  for (int t = 0; t < 4; ++t) b_cur[t] = *(const v16bf*)(bptr[t]);

  int kk = 0;
  for (; kk < K - 32; kk += 32) {
    v16bf a_nxt = load_a_frag(aptr + kk + 32, lane);
    v16bf b_nxt[4];
#pragma unroll
    for (int t = 0; t < 4; ++t) b_nxt[t] = *(const v16bf*)(bptr[t] + kk + 32);
#pragma unroll
    for (int t = 0; t < 4; ++t) c[t] = wmma_bf16(a_cur, b_cur[t], c[t]);
    a_cur = a_nxt;
#pragma unroll
    for (int t = 0; t < 4; ++t) b_cur[t] = b_nxt[t];
  }
#pragma unroll
  for (int t = 0; t < 4; ++t) c[t] = wmma_bf16(a_cur, b_cur[t], c[t]);

#pragma unroll
  for (int t = 0; t < 4; ++t) {
    const int col = col0 + t * 16 + (lane & 15);
    const float bv = bias[col];
#pragma unroll
    for (int r = 0; r < 8; ++r) {
      const int row = row0 + r + ((lane >> 4) << 3);
      const float v = c[t][r] + bv;
      if (OUT_BF16) ((__bf16*)Out)[(size_t)row * N + col] = f2bf(v);
      else          ((float*)Out)[(size_t)row * N + col] = v;
    }
  }
}

// Vt[b][d][r] = Vproj[b][r>>3][(r&7)*128 + d]   (r = s*8+h over 8192, d over 128)
__global__ void vtrans(const __bf16* __restrict__ Vp, __bf16* __restrict__ Vt) {
  size_t t = (size_t)blockIdx.x * blockDim.x + threadIdx.x;   // total 2*128*8192
  const int r = (int)(t & (SH - 1));
  const int d = (int)((t >> 13) & (HD - 1));
  const int b = (int)(t >> 20);
  const int s = r >> 3, hh = r & 7;
  Vt[t] = Vp[((size_t)b * Ssz + s) * Esz + hh * HD + d];
}

// score[b, h*1024+q, n] = (1/sqrt(128)) * sum_k Q[b,q,h*128+k] * K[b, n>>3, (n&7)*128+k]
__global__ void __launch_bounds__(256)
score_gemm(const __bf16* __restrict__ Qp, const __bf16* __restrict__ Kp,
           float* __restrict__ Wt) {
  const int lane = threadIdx.x & 31;
  const int wave = threadIdx.x >> 5;
  const int b = blockIdx.z;
  const int row0 = blockIdx.y * 128 + wave * 16;   // r = h*1024 + q
  const int col0 = blockIdx.x * 64;

  const int r = row0 + (lane & 15);
  const int h = r >> 10, q = r & 1023;
  const __bf16* aptr = Qp + ((size_t)b * Ssz + q) * Esz + h * HD;

  const __bf16* bptr[4];
#pragma unroll
  for (int t = 0; t < 4; ++t) {
    const int n = col0 + t * 16 + (lane & 15);
    bptr[t] = Kp + ((size_t)b * Ssz + (n >> 3)) * Esz + (n & 7) * HD +
              ((lane >> 4) << 4);
  }

  v8f c[4] = {};
  v16bf a_cur = load_a_frag(aptr, lane);
  v16bf b_cur[4];
#pragma unroll
  for (int t = 0; t < 4; ++t) b_cur[t] = *(const v16bf*)(bptr[t]);

#pragma unroll
  for (int kk = 0; kk < HD - 32; kk += 32) {
    v16bf a_nxt = load_a_frag(aptr + kk + 32, lane);
    v16bf b_nxt[4];
#pragma unroll
    for (int t = 0; t < 4; ++t) b_nxt[t] = *(const v16bf*)(bptr[t] + kk + 32);
#pragma unroll
    for (int t = 0; t < 4; ++t) c[t] = wmma_bf16(a_cur, b_cur[t], c[t]);
    a_cur = a_nxt;
#pragma unroll
    for (int t = 0; t < 4; ++t) b_cur[t] = b_nxt[t];
  }
#pragma unroll
  for (int t = 0; t < 4; ++t) c[t] = wmma_bf16(a_cur, b_cur[t], c[t]);

  const float scale = 0.08838834764831845f;        // 1/sqrt(128)
  float* wb = Wt + (size_t)b * ((size_t)Hsz * Ssz * SH);
#pragma unroll
  for (int t = 0; t < 4; ++t) {
    const int col = col0 + t * 16 + (lane & 15);
#pragma unroll
    for (int rr = 0; rr < 8; ++rr) {
      const int row = row0 + rr + ((lane >> 4) << 3);
      wb[(size_t)row * SH + col] = c[t][rr] * scale;
    }
  }
}

// in-place softmax over the h axis (8 values at stride 1024*8192 floats).
__global__ void softmax_h(float* __restrict__ Wt, __bf16* __restrict__ Wtb) {
  size_t t = (size_t)blockIdx.x * blockDim.x + threadIdx.x;  // total 2*1024*8192
  const int n = (int)(t & (SH - 1));
  const int q = (int)((t >> 13) & (Ssz - 1));
  const int b = (int)(t >> 23);
  const size_t off = (size_t)b * ((size_t)Hsz * Ssz * SH) + (size_t)q * SH + n;
  float* base = Wt + off;
  const size_t hs = (size_t)Ssz * SH;
  float x[Hsz];
#pragma unroll
  for (int h = 0; h < Hsz; ++h) x[h] = base[h * hs];
  float m = x[0];
#pragma unroll
  for (int h = 1; h < Hsz; ++h) m = fmaxf(m, x[h]);
  float sum = 0.f;
#pragma unroll
  for (int h = 0; h < Hsz; ++h) { x[h] = __expf(x[h] - m); sum += x[h]; }
  const float inv = 1.0f / sum;
  if (Wtb) {
    __bf16* bb = Wtb + off;
#pragma unroll
    for (int h = 0; h < Hsz; ++h) {
      const float v = x[h] * inv;
      base[h * hs] = v;
      bb[h * hs] = f2bf(v);
    }
  } else {
#pragma unroll
    for (int h = 0; h < Hsz; ++h) base[h * hs] = x[h] * inv;
  }
}

// PV GEMM with TDM double-buffered A staging in LDS.
// Block = 4 waves = 64 q-rows for one (b,h); wave -> 16 rows x 128 cols.
// Wave 0 issues tensor_load_to_lds 2 chunks ahead; TENSORcnt + barrier sync.
__global__ void __launch_bounds__(128)
pv_gemm_tdm(const __bf16* __restrict__ Wtb, const __bf16* __restrict__ Vt,
            __bf16* __restrict__ Obf) {
  __shared__ __bf16 Abuf[2][PV_ROWS * PV_LROW];
  const int lane = threadIdx.x & 31;
  const int wave = threadIdx.x >> 5;
  const int b = blockIdx.z, h = blockIdx.y;
  const int row0 = blockIdx.x * PV_ROWS;          // q-row base of this block
  const __bf16* Abase = Wtb + (((size_t)b * Hsz + h) * Ssz + row0) * SH;
  const __bf16* bbase = Vt + (size_t)b * HD * SH;
  const __bf16* bptr[8];
#pragma unroll
  for (int t = 0; t < 8; ++t)
    bptr[t] = bbase + (size_t)(t * 16 + (lane & 15)) * SH + ((lane >> 4) << 4);

  const int NCH = SH / PV_KCH;                    // 64 chunks
  if (wave == 0) {
    tdm_load_a((unsigned int)(uintptr_t)&Abuf[0][0], Abase);
    tdm_load_a((unsigned int)(uintptr_t)&Abuf[1][0], Abase + PV_KCH);
  }

  v8f c[8] = {};
  const int rloc = wave * 16 + (lane & 15);
  for (int ch = 0; ch < NCH; ++ch) {
    if (wave == 0) {
      if (ch + 1 < NCH) __builtin_amdgcn_s_wait_tensorcnt(1);
      else              __builtin_amdgcn_s_wait_tensorcnt(0);
    }
    __syncthreads();                              // chunk `ch` visible to all
    const __bf16* ab = &Abuf[ch & 1][(size_t)rloc * PV_LROW];
#pragma unroll
    for (int kl = 0; kl < PV_KCH; kl += 32) {
      v16bf a = load_a_frag(ab + kl, lane);       // ds_load, conflict-free
      const int kg = ch * PV_KCH + kl;
#pragma unroll
      for (int t = 0; t < 8; ++t)
        c[t] = wmma_bf16(a, *(const v16bf*)(bptr[t] + kg), c[t]);
    }
    __syncthreads();                              // all done reading buf[ch&1]
    if (wave == 0 && ch + 2 < NCH)
      tdm_load_a((unsigned int)(uintptr_t)&Abuf[ch & 1][0],
                 Abase + (size_t)(ch + 2) * PV_KCH);
  }

#pragma unroll
  for (int t = 0; t < 8; ++t) {
    const int d = t * 16 + (lane & 15);
#pragma unroll
    for (int rr = 0; rr < 8; ++rr) {
      const int q = row0 + wave * 16 + rr + ((lane >> 4) << 3);
      Obf[((size_t)b * Ssz + q) * Esz + h * HD + d] = f2bf(c[t][rr]);
    }
  }
}

// Fallback PV GEMM (no bf16 weight copy): A from fp32, converted in-register.
__global__ void __launch_bounds__(256)
pv_gemm_f32(const float* __restrict__ Wt, const __bf16* __restrict__ Vt,
            __bf16* __restrict__ Obf) {
  const int lane = threadIdx.x & 31;
  const int wave = threadIdx.x >> 5;
  const int b = blockIdx.z, h = blockIdx.y;
  const int row0 = blockIdx.x * 128 + wave * 16;
  const float* aptr =
      Wt + (((size_t)b * Hsz + h) * Ssz + (row0 + (lane & 15))) * SH;
  const __bf16* bbase = Vt + (size_t)b * HD * SH;
  const __bf16* bptr[8];
#pragma unroll
  for (int t = 0; t < 8; ++t)
    bptr[t] = bbase + (size_t)(t * 16 + (lane & 15)) * SH + ((lane >> 4) << 4);

  v8f c[8] = {};
  v16bf a_cur = load_a_frag_f32(aptr, lane);
  int kk = 0;
  for (; kk < SH - 32; kk += 32) {
    v16bf a_nxt = load_a_frag_f32(aptr + kk + 32, lane);
    v16bf bfr[8];
#pragma unroll
    for (int t = 0; t < 8; ++t) bfr[t] = *(const v16bf*)(bptr[t] + kk);
#pragma unroll
    for (int t = 0; t < 8; ++t) c[t] = wmma_bf16(a_cur, bfr[t], c[t]);
    a_cur = a_nxt;
  }
  {
    v16bf bfr[8];
#pragma unroll
    for (int t = 0; t < 8; ++t) bfr[t] = *(const v16bf*)(bptr[t] + kk);
#pragma unroll
    for (int t = 0; t < 8; ++t) c[t] = wmma_bf16(a_cur, bfr[t], c[t]);
  }
#pragma unroll
  for (int t = 0; t < 8; ++t) {
    const int d = t * 16 + (lane & 15);
#pragma unroll
    for (int rr = 0; rr < 8; ++rr) {
      const int q = row0 + rr + ((lane >> 4) << 3);
      Obf[((size_t)b * Ssz + q) * Esz + h * HD + d] = f2bf(c[t][rr]);
    }
  }
}

// ---------- launcher ----------
extern "C" void kernel_launch(void* const* d_in, const int* in_sizes, int n_in,
                              void* d_out, int out_size, void* d_ws, size_t ws_size,
                              hipStream_t stream) {
  const float* query = (const float*)d_in[0];
  const float* key_  = (const float*)d_in[1];
  const float* value = (const float*)d_in[2];
  const float* Wq = (const float*)d_in[3];  const float* bq = (const float*)d_in[4];
  const float* Wk = (const float*)d_in[5];  const float* bk = (const float*)d_in[6];
  const float* Wv = (const float*)d_in[7];  const float* bv = (const float*)d_in[8];
  const float* Wo = (const float*)d_in[9];  const float* bo = (const float*)d_in[10];

  float* out_f = (float*)d_out;                                   // [2,1024,1024]
  float* weight = out_f + (size_t)Bsz * Ssz * Esz;                // [2,8,1024,8192]

  char* w = (char*)d_ws;
  auto carve = [&](size_t bytes) {
    void* p = (void*)w; w += (bytes + 255) & ~(size_t)255; return p;
  };
  const size_t XE = (size_t)RTOT * Esz;       // 2048*1024
  const size_t WE = (size_t)Esz * Esz;        // 1024*1024
  __bf16* qx = (__bf16*)carve(XE * 2);
  __bf16* kx = (__bf16*)carve(XE * 2);
  __bf16* vx = (__bf16*)carve(XE * 2);
  __bf16* wqb = (__bf16*)carve(WE * 2);
  __bf16* wkb = (__bf16*)carve(WE * 2);
  __bf16* wvb = (__bf16*)carve(WE * 2);
  __bf16* wob = (__bf16*)carve(WE * 2);
  __bf16* Qp = (__bf16*)carve(XE * 2);
  __bf16* Kp = (__bf16*)carve(XE * 2);
  __bf16* Vp = (__bf16*)carve(XE * 2);
  __bf16* Vt = (__bf16*)carve((size_t)Bsz * HD * SH * 2);
  __bf16* Ob = (__bf16*)carve(XE * 2);

  const size_t wtb_bytes = (size_t)Bsz * Hsz * Ssz * SH * 2;  // 256 MB
  __bf16* Wtb = nullptr;
  if ((size_t)(w - (char*)d_ws) + wtb_bytes <= ws_size)
    Wtb = (__bf16*)carve(wtb_bytes);

  const int TB = 256;
  cvt_f32_bf16<<<(unsigned)((XE / 8 + TB - 1) / TB), TB, 0, stream>>>(query, qx, XE / 8);
  cvt_f32_bf16<<<(unsigned)((XE / 8 + TB - 1) / TB), TB, 0, stream>>>(key_,  kx, XE / 8);
  cvt_f32_bf16<<<(unsigned)((XE / 8 + TB - 1) / TB), TB, 0, stream>>>(value, vx, XE / 8);
  cvt_f32_bf16<<<(unsigned)((WE / 8 + TB - 1) / TB), TB, 0, stream>>>(Wq, wqb, WE / 8);
  cvt_f32_bf16<<<(unsigned)((WE / 8 + TB - 1) / TB), TB, 0, stream>>>(Wk, wkb, WE / 8);
  cvt_f32_bf16<<<(unsigned)((WE / 8 + TB - 1) / TB), TB, 0, stream>>>(Wv, wvb, WE / 8);
  cvt_f32_bf16<<<(unsigned)((WE / 8 + TB - 1) / TB), TB, 0, stream>>>(Wo, wob, WE / 8);

  dim3 gproj(Esz / 64, RTOT / 128, 1);
  gemm_xwT<true><<<gproj, TB, 0, stream>>>(qx, wqb, bq, Qp, RTOT, Esz, Esz);
  gemm_xwT<true><<<gproj, TB, 0, stream>>>(kx, wkb, bk, Kp, RTOT, Esz, Esz);
  gemm_xwT<true><<<gproj, TB, 0, stream>>>(vx, wvb, bv, Vp, RTOT, Esz, Esz);

  {
    const size_t n = (size_t)Bsz * HD * SH;
    vtrans<<<(unsigned)(n / TB), TB, 0, stream>>>(Vp, Vt);
  }

  score_gemm<<<dim3(SH / 64, SH / 128, Bsz), TB, 0, stream>>>(Qp, Kp, weight);

  {
    const size_t n = (size_t)Bsz * Ssz * SH;   // 2^24
    softmax_h<<<(unsigned)(n / TB), TB, 0, stream>>>(weight, Wtb);
  }

  if (Wtb)
    pv_gemm_tdm<<<dim3(Ssz / PV_ROWS, Hsz, Bsz), 128, 0, stream>>>(Wtb, Vt, Ob);
  else
    pv_gemm_f32<<<dim3(Ssz / 128, Hsz, Bsz), TB, 0, stream>>>(weight, Vt, Ob);

  gemm_xwT<false><<<gproj, TB, 0, stream>>>(Ob, wob, bo, out_f, RTOT, Esz, Esz);
}
